// DecoderPooler_87883620811288
// MI455X (gfx1250) — compile-verified, matrix operand order checked by reading, not run on hardware
//
#include <hip/hip_runtime.h>

// DecoderPooler on gfx1250 (MI455X):
//   out[b,:] = hidden_state[b, sum(mask[b,:])-1, :]
//
// Kernel 1: one wave32 computes all 16 row sums of the 16x4096 int32 mask
//           with V_WMMA_I32_16X16X64_IU8 (A = mask bytes, B = all-ones u8),
//           64 accumulating WMMAs cover K=4096. Indices -> d_ws.
// Kernel 2: 16 blocks x 256 threads gather one H=1024 float row per batch
//           as float4 (global_load_b128 / global_store_b128).

typedef __attribute__((ext_vector_type(8))) int v8i;

constexpr int Bn = 16;    // batch
constexpr int S  = 4096;  // sequence length
constexpr int H  = 1024;  // hidden

// ---------------------------------------------------------------------------
// Kernel 1: mask row-sums via IU8 WMMA, one full wave32 (EXEC all ones).
// ---------------------------------------------------------------------------
__global__ __launch_bounds__(32) void pool_idx_wmma(const int* __restrict__ mask,
                                                    int* __restrict__ last_idx) {
  const int lane = threadIdx.x;   // 0..31
  const int m    = lane & 15;     // A-matrix row == batch index
  const int half = lane >> 4;     // 0: lanes 0-15, 1: lanes 16-31

  v8i acc = {};                   // 16x16 i32 C/D accumulator (8 VGPRs)

  v8i bones;                      // B = 64x16 all-ones u8: layout-independent
#pragma unroll
  for (int j = 0; j < 8; ++j) bones[j] = 0x01010101;

  const int* __restrict__ rowp = mask + m * S;

  // 8-bit A 16x64 layout: VGPR v (g=v>>1, w=v&1), lane half h:
  //   K = it*64 + g*16 + h*8 + w*4 .. +3
  for (int it = 0; it < S / 64; ++it) {
    v8i a;
#pragma unroll
    for (int v = 0; v < 8; ++v) {
      const int g = v >> 1, w = v & 1;
      const int kbase = it * 64 + g * 16 + half * 8 + w * 4;
      const int4 q = *reinterpret_cast<const int4*>(rowp + kbase);  // b128 load
      // pack four 0/1 int32 mask values into one byte-packed dword
      a[v] = (q.x & 1) | ((q.y & 1) << 8) | ((q.z & 1) << 16) | ((q.w & 1) << 24);
    }
    // D[m][n] += sum_k A[m][k] * 1  -> every column of row m holds the row sum
    acc = __builtin_amdgcn_wmma_i32_16x16x64_iu8(
        /*sgn_a=*/false, a, /*sgn_b=*/false, bones, acc,
        /*reuse_a=*/false, /*reuse_b=*/false);
  }

  // 16x16 i32 D layout: VGPR j, lanes 0-15 -> D[M=j][N=lane];
  //                              lanes 16-31 -> D[M=8+j][N=lane-16]
  if (lane == 0) {
#pragma unroll
    for (int j = 0; j < 8; ++j) last_idx[j] = acc[j] - 1;
  } else if (lane == 16) {
#pragma unroll
    for (int j = 0; j < 8; ++j) last_idx[8 + j] = acc[j] - 1;
  }
}

// ---------------------------------------------------------------------------
// Kernel 2: gather selected rows, fully vectorized (one float4 per thread).
// ---------------------------------------------------------------------------
__global__ __launch_bounds__(256) void gather_rows(const float* __restrict__ hs,
                                                   const int* __restrict__ last_idx,
                                                   float* __restrict__ out) {
  const int b = blockIdx.x;       // batch
  const int t = threadIdx.x;      // 0..255 -> H/4 float4s
  const int row = last_idx[b];    // scalar (uniform) load

  const float4* __restrict__ src = reinterpret_cast<const float4*>(
      hs + (size_t)b * S * H + (size_t)row * H);
  float4* __restrict__ dst = reinterpret_cast<float4*>(out + (size_t)b * H);
  dst[t] = src[t];
}

// ---------------------------------------------------------------------------
extern "C" void kernel_launch(void* const* d_in, const int* in_sizes, int n_in,
                              void* d_out, int out_size, void* d_ws, size_t ws_size,
                              hipStream_t stream) {
  const float* hidden = (const float*)d_in[0];   // [B,S,H] f32
  const int*   mask   = (const int*)d_in[1];     // [B,S] i32 (contiguous prefix)
  float*       out    = (float*)d_out;           // [B,H] f32
  int*         idx    = (int*)d_ws;              // 16 ints of scratch

  (void)in_sizes; (void)n_in; (void)out_size; (void)ws_size;

  pool_idx_wmma<<<1, 32, 0, stream>>>(mask, idx);
  gather_rows<<<Bn, 256, 0, stream>>>(hidden, idx, out);
}
